// RetinaFace_Biubug6_64974265254171
// MI455X (gfx1250) — matrix operand order repaired and checked by reading, not compiled, not run on hardware
//
#include <hip/hip_runtime.h>
#include <hip/hip_bf16.h>
#include <math.h>

typedef __attribute__((ext_vector_type(16))) _Float16 v16h;
typedef __attribute__((ext_vector_type(8)))  _Float16 v8h;
typedef __attribute__((ext_vector_type(8)))  float    v8f;

#define NTOT      43008
#define NSORT     65536
#define TOPK      1000
#define SCORE_THR 0.02f
#define IOU_THR   0.4f
#define BATCH     8

// ---------------- f32 -> f16 convert ----------------
__global__ void k_cvt_f16(const float* __restrict__ in, _Float16* __restrict__ out, int n) {
  int i = blockIdx.x * blockDim.x + threadIdx.x;
  if (i < n) out[i] = (_Float16)in[i];
}

// ---------------- weight pack + K-reorder ----------------
// w (OIHW flat): w[n][ci][dy][dx] = w[n*Cin*9 + ci*9 + (dy*3+dx)]
// out[n*KPAD + k] with k = s*Cin + ci (s = dy*3+dx, channel-innermost), zero-padded to KPAD.
__global__ void k_pack_w(const float* __restrict__ w, _Float16* __restrict__ outB,
                         int CIN, int Cout, int KPAD) {
  int i = blockIdx.x * blockDim.x + threadIdx.x;
  if (i >= Cout * KPAD) return;
  int n = i / KPAD, k = i - n * KPAD;
  int K = CIN * 9;
  _Float16 v = (_Float16)0.0f;
  if (k < K) {
    int s = k / CIN, ci = k - s * CIN;
    v = (_Float16)w[n * K + ci * 9 + s];
  }
  outB[i] = v;
}

// ---------------- implicit-GEMM 3x3 conv via WMMA f16 ----------------
// K order: k = (dy*3+dx)*CIN + ci  -> every 8-element K-run = 8 consecutive channels at one
// spatial tap -> one aligned ds_load_b128. Input halo tile staged in LDS (zero borders),
// channel-innermost with XOR-by-8 bank swizzle. Weights staged as [16][KSTRIDE] panels.
template <int CIN, int WH>
__global__ void __launch_bounds__(256) k_conv3x3_wmma(
    const _Float16* __restrict__ in,    // [B, CIN, WH, WH]
    const _Float16* __restrict__ wgt,   // [Cout, KPAD] f16, K-reordered
    const float*    __restrict__ bias,  // [Cout]
    _Float16*       __restrict__ out,   // [B, CoutTotal, WH, WH]
    int Cout, int CoutTotal, int coutOff, int act) // act 0:none 1:relu 2:leaky.1
{
  constexpr int K       = CIN * 9;
  constexpr int KPAD    = (K + 31) & ~31;
  constexpr int KSTRIDE = KPAD + 8;                 // conflict-free B rows
  constexpr int lgCIN   = (CIN == 64) ? 6 : 4;
  constexpr int lgW     = (WH == 128) ? 7 : ((WH == 64) ? 6 : 5);
  constexpr int HW      = WH * WH;
  constexpr int NR      = 128 / WH;                 // output rows per block strip
  constexpr int RWS     = NR + 2;                   // staged rows (with halo)
  constexpr int WP      = WH + 2;                   // staged cols (with halo)
  constexpr int SWM     = CIN / 8 - 1;              // swizzle mask

  __shared__ _Float16 Xs[RWS * WP * CIN];
  __shared__ _Float16 Bs[16 * KSTRIDE];

  const int coutBase = blockIdx.y * 16;
  const int b        = blockIdx.z;
  const int tile0    = blockIdx.x * 128;
  const int ybase    = tile0 >> lgW;

  // ---- stage weight panel: straight coalesced copy (already reordered+padded) ----
  for (int idx = threadIdx.x; idx < 16 * KPAD; idx += 256) {
    int n = idx / KPAD, k = idx - n * KPAD;
    Bs[n * KSTRIDE + k] = wgt[(size_t)(coutBase + n) * KPAD + k];
  }

  // ---- stage input halo tile with zero borders ----
  const _Float16* inB = in + (size_t)b * CIN * HW;
  for (int idx = threadIdx.x; idx < CIN * RWS * WP; idx += 256) {
    int ci  = idx / (RWS * WP);
    int rem = idx - ci * (RWS * WP);
    int r   = rem / WP;
    int c   = rem - r * WP;
    int y   = ybase - 1 + r;
    int x   = c - 1;
    _Float16 val = (_Float16)0.0f;
    if ((unsigned)y < (unsigned)WH && (unsigned)x < (unsigned)WH)
      val = inB[((size_t)ci << (2 * lgW)) + (y << lgW) + x];
    Xs[(r * WP + c) * CIN + (ci ^ ((c & SWM) << 3))] = val;
  }
  __syncthreads();

  const int wave = threadIdx.x >> 5;
  const int lane = threadIdx.x & 31;
  const int hh   = lane >> 4;
  const int m    = lane & 15;

  const int tileBase = tile0 + wave * 16;
  const int p    = tileBase + m;
  const int yloc = p >> lgW;
  const int x    = p & (WH - 1);
  const int rloc = yloc - ybase + 1;

  auto loadA = [&](int run) -> v8h {
    int s   = run >> lgCIN;                 // spatial tap 0..8
    int ci0 = run & (CIN - 1);              // first of 8 consecutive channels
    int dy  = s / 3;
    int dx  = s - 3 * dy;
    int row = rloc + dy - 1;
    int cc  = x + dx;                       // halo column
    int off = (row * WP + cc) * CIN + (ci0 ^ ((cc & SWM) << 3));
    return *(const v8h*)&Xs[off];
  };

  v8f acc = {};
  for (int k0 = 0; k0 < K; k0 += 32) {
    int run1 = k0 + hh * 8;
    int run2 = k0 + 16 + hh * 8;
    v8h alo = loadA(run1);
    v8h ahi;
    if constexpr (K % 32 != 0) {            // tail K-block: upper run may be past K
      ahi = (run2 < K) ? loadA(run2) : (v8h){};
    } else {
      ahi = loadA(run2);
    }
    v16h a = __builtin_shufflevector(alo, ahi, 0,1,2,3,4,5,6,7,8,9,10,11,12,13,14,15);

    const v8h* bp = (const v8h*)&Bs[m * KSTRIDE + k0 + hh * 8];
    v16h bfv = __builtin_shufflevector(bp[0], bp[2], 0,1,2,3,4,5,6,7,8,9,10,11,12,13,14,15);

    acc = __builtin_amdgcn_wmma_f32_16x16x32_f16(false, a, false, bfv, (short)0, acc, false, false);
  }

  // C layout: VGPR r -> M = r (lanes 0-15) / 8+r (lanes 16-31); N = lane&15.
  const int c = coutBase + m;
  float bb = bias[c];
  v8h o;
#pragma unroll
  for (int r = 0; r < 8; ++r) {
    float val = acc[r] + bb;
    if (act == 1)      val = val > 0.0f ? val : 0.0f;
    else if (act == 2) val = val > 0.0f ? val : 0.1f * val;
    o[r] = (_Float16)val;
  }
  _Float16* op = out + ((size_t)b * CoutTotal + coutOff + c) * HW + tileBase + hh * 8;
  *(v8h*)op = o;   // 8 contiguous pixels, 16B aligned
}

// ---------------- fused 1x1 heads + softmax + prior decode ----------------
__global__ void k_head_decode(
    const _Float16* __restrict__ ssh,  // [B, 64, H*W]
    const float* __restrict__ wB, const float* __restrict__ bB,  // [8][64], [8]
    const float* __restrict__ wC, const float* __restrict__ bC,  // [4][64], [4]
    float* __restrict__ boxes,   // [B, NTOT, 4]
    float* __restrict__ scores,  // [B, NTOT]
    int H, int W, int levelBase, float stride, float size0, float size1)
{
  int gid = blockIdx.x * blockDim.x + threadIdx.x;
  int HW = H * W;
  if (gid >= BATCH * HW) return;
  int b = gid / HW, p = gid - b * HW;
  int y0 = p / W, x0 = p - y0 * W;

  float reg[8], cls[4];
#pragma unroll
  for (int o = 0; o < 8; ++o) reg[o] = bB[o];
#pragma unroll
  for (int o = 0; o < 4; ++o) cls[o] = bC[o];

  const _Float16* xp = ssh + (size_t)b * 64 * HW + p;
  for (int c = 0; c < 64; ++c) {
    float xv = (float)xp[(size_t)c * HW];
#pragma unroll
    for (int o = 0; o < 8; ++o) reg[o] += wB[o * 64 + c] * xv;
#pragma unroll
    for (int o = 0; o < 4; ++o) cls[o] += wC[o * 64 + c] * xv;
  }

  const float inv = 1.0f / 1024.0f;
  float pcx = (x0 + 0.5f) * stride * inv;
  float pcy = (y0 + 0.5f) * stride * inv;
  float sz[2] = { size0 * inv, size1 * inv };
#pragma unroll
  for (int a2 = 0; a2 < 2; ++a2) {
    float s  = 1.0f / (1.0f + __expf(cls[a2 * 2] - cls[a2 * 2 + 1]));
    float cx = pcx + reg[a2 * 4 + 0] * 0.1f * sz[a2];
    float cy = pcy + reg[a2 * 4 + 1] * 0.1f * sz[a2];
    float bw = sz[a2] * __expf(reg[a2 * 4 + 2] * 0.2f);
    float bh = sz[a2] * __expf(reg[a2 * 4 + 3] * 0.2f);
    int nidx = levelBase + p * 2 + a2;
    float* bp = boxes + ((size_t)b * NTOT + nidx) * 4;
    bp[0] = cx - 0.5f * bw; bp[1] = cy - 0.5f * bh;
    bp[2] = cx + 0.5f * bw; bp[3] = cy + 0.5f * bh;
    scores[(size_t)b * NTOT + nidx] = s;
  }
}

// ---------------- sortable keys ----------------
__global__ void k_keys(const float* __restrict__ scores, unsigned long long* __restrict__ keys) {
  int gid = blockIdx.x * blockDim.x + threadIdx.x;
  if (gid >= BATCH * NSORT) return;
  int b = gid / NSORT, i = gid - b * NSORT;
  unsigned long long key = 0ULL;
  if (i < NTOT) {
    float s = scores[(size_t)b * NTOT + i];
    unsigned bits = (s > SCORE_THR) ? __float_as_uint(s) : 0u;
    key = ((unsigned long long)bits << 32) | (unsigned)(0xFFFFFFFFu - (unsigned)i);
  }
  keys[gid] = key;
}

// ---------------- one bitonic pass (descending) ----------------
__global__ void k_bitonic(unsigned long long* __restrict__ keys, int k, int j) {
  int gid = blockIdx.x * blockDim.x + threadIdx.x;
  if (gid >= BATCH * NSORT) return;
  int b = gid / NSORT, i = gid - b * NSORT;
  int l = i ^ j;
  if (l <= i) return;
  unsigned long long* base = keys + (size_t)b * NSORT;
  unsigned long long a = base[i], c = base[l];
  bool up = ((i & k) == 0);
  bool sw = up ? (a < c) : (a > c);
  if (sw) { base[i] = c; base[l] = a; }
}

// ---------------- greedy NMS, one 1024-thread block per batch ----------------
__global__ void __launch_bounds__(1024) k_nms(
    const unsigned long long* __restrict__ keys,
    const float* __restrict__ boxes,
    float* __restrict__ outp)
{
  __shared__ float bx0[TOPK], by0[TOPK], bx1[TOPK], by1[TOPK], sc[TOPK];
  __shared__ int   keep[TOPK];
  int b = blockIdx.x, tid = threadIdx.x;

  for (int t = tid; t < TOPK; t += 1024) {
    unsigned long long key = keys[(size_t)b * NSORT + t];
    unsigned idx = 0xFFFFFFFFu - (unsigned)(key & 0xFFFFFFFFull);
    float s = __uint_as_float((unsigned)(key >> 32));
    if (idx >= (unsigned)NTOT) { idx = 0; s = 0.0f; }
    const float* bp = boxes + ((size_t)b * NTOT + idx) * 4;
    bx0[t] = bp[0]; by0[t] = bp[1]; bx1[t] = bp[2]; by1[t] = bp[3];
    sc[t] = s; keep[t] = (s > SCORE_THR) ? 1 : 0;
  }
  __syncthreads();

  for (int i = 0; i < TOPK; ++i) {
    if (keep[i] && tid > i && tid < TOPK && keep[tid]) {
      float ax0 = bx0[i], ay0 = by0[i], ax1 = bx1[i], ay1 = by1[i];
      float areaA = (ax1 - ax0) * (ay1 - ay0);
      float areaB = (bx1[tid] - bx0[tid]) * (by1[tid] - by0[tid]);
      float cx0 = fmaxf(ax0, bx0[tid]), cy0 = fmaxf(ay0, by0[tid]);
      float cx1 = fminf(ax1, bx1[tid]), cy1 = fminf(ay1, by1[tid]);
      float iw = fmaxf(0.0f, cx1 - cx0), ih = fmaxf(0.0f, cy1 - cy0);
      float inter = iw * ih;
      float iou = inter / (areaA + areaB - inter + 1e-9f);
      if (iou > IOU_THR) keep[tid] = 0;
    }
    __syncthreads();
  }

  for (int t = tid; t < TOPK; t += 1024) {
    float* op = outp + ((size_t)b * TOPK + t) * 5;
    op[0] = bx0[t]; op[1] = by0[t]; op[2] = bx1[t]; op[3] = by1[t];
    op[4] = sc[t] * (keep[t] ? 1.0f : 0.0f);
  }
}

// ---------------- per-level conv chain (templated on spatial size) ----------------
template <int WH>
static void run_level_convs(const _Float16* xh, _Float16* const* pw, const float* const* sshB,
                            _Float16* so, _Float16* tA, _Float16* tB, hipStream_t stream) {
  constexpr int HW = WH * WH;
  dim3 blk(256);
  dim3 gC1(HW / 128, 2, BATCH);  // Cout=32 -> 2 N-tiles
  dim3 gC (HW / 128, 1, BATCH);  // Cout=16 -> 1 N-tile
  k_conv3x3_wmma<64, WH><<<gC1, blk, 0, stream>>>(xh, pw[0], sshB[0], so, 32, 64, 0,  1); // c1
  k_conv3x3_wmma<64, WH><<<gC,  blk, 0, stream>>>(xh, pw[1], sshB[1], tA, 16, 16, 0,  2); // c2
  k_conv3x3_wmma<16, WH><<<gC,  blk, 0, stream>>>(tA, pw[2], sshB[2], so, 16, 64, 32, 1); // c3
  k_conv3x3_wmma<16, WH><<<gC,  blk, 0, stream>>>(tA, pw[3], sshB[3], tB, 16, 16, 0,  2); // c4
  k_conv3x3_wmma<16, WH><<<gC,  blk, 0, stream>>>(tB, pw[4], sshB[4], so, 16, 64, 48, 1); // c5
}

// =====================================================================
extern "C" void kernel_launch(void* const* d_in, const int* in_sizes, int n_in,
                              void* d_out, int out_size, void* d_ws, size_t ws_size,
                              hipStream_t stream) {
  (void)in_sizes; (void)n_in; (void)out_size; (void)ws_size;

  // Input order (jax dict flatten, sorted keys): p3, p4, p5,
  // params.box[0..2]{b,w}, params.cls[0..2]{b,w}, params.ssh[0..2]{c1..c5}{b,w}
  const float* pIn[3] = { (const float*)d_in[0], (const float*)d_in[1], (const float*)d_in[2] };
  const float *boxB[3], *boxW[3], *clsB[3], *clsW[3], *sshB[3][5], *sshW[3][5];
  for (int l = 0; l < 3; ++l) {
    boxB[l] = (const float*)d_in[3 + 2 * l];  boxW[l] = (const float*)d_in[4 + 2 * l];
    clsB[l] = (const float*)d_in[9 + 2 * l];  clsW[l] = (const float*)d_in[10 + 2 * l];
    for (int j = 0; j < 5; ++j) {
      sshB[l][j] = (const float*)d_in[15 + 10 * l + 2 * j];
      sshW[l][j] = (const float*)d_in[16 + 10 * l + 2 * j];
    }
  }

  const int   HWs[3] = {16384, 4096, 1024};
  const float strides[3] = {8.f, 16.f, 32.f};
  const float sz0[3] = {16.f, 64.f, 256.f}, sz1[3] = {32.f, 128.f, 512.f};
  const int   lvlBase[3] = {0, 32768, 40960};
  const int   CinC[5]  = {64, 64, 16, 16, 16};
  const int   Co[5]    = {32, 16, 16, 16, 16};
  const int   KPadC[5] = {576, 576, 160, 160, 160};

  // ---- workspace carve-up ----
  char* ws = (char*)d_ws;
  size_t off = 0;
  auto carve = [&](size_t bytes) -> char* {
    char* q = ws + off;
    off += (bytes + 255) & ~(size_t)255;
    return q;
  };
  _Float16 *xh[3], *so[3];
  for (int l = 0; l < 3; ++l) xh[l] = (_Float16*)carve((size_t)BATCH * 64 * HWs[l] * 2);
  for (int l = 0; l < 3; ++l) so[l] = (_Float16*)carve((size_t)BATCH * 64 * HWs[l] * 2);
  _Float16* tA = (_Float16*)carve((size_t)BATCH * 16 * HWs[0] * 2);
  _Float16* tB = (_Float16*)carve((size_t)BATCH * 16 * HWs[0] * 2);
  _Float16* pw[3][5];
  for (int l = 0; l < 3; ++l)
    for (int j = 0; j < 5; ++j) pw[l][j] = (_Float16*)carve((size_t)Co[j] * KPadC[j] * 2);
  float* boxes  = (float*)carve((size_t)BATCH * NTOT * 4 * 4);
  float* scores = (float*)carve((size_t)BATCH * NTOT * 4);
  unsigned long long* keys = (unsigned long long*)carve((size_t)BATCH * NSORT * 8);

  // ---- convert activations to f16; pack+reorder weights ----
  for (int l = 0; l < 3; ++l) {
    int n = BATCH * 64 * HWs[l];
    k_cvt_f16<<<(n + 255) / 256, 256, 0, stream>>>(pIn[l], xh[l], n);
    for (int j = 0; j < 5; ++j) {
      int m = Co[j] * KPadC[j];
      k_pack_w<<<(m + 255) / 256, 256, 0, stream>>>(sshW[l][j], pw[l][j], CinC[j], Co[j], KPadC[j]);
    }
  }

  // ---- SSH conv chains + fused heads ----
  for (int l = 0; l < 3; ++l) {
    if (l == 0) run_level_convs<128>(xh[l], pw[l], sshB[l], so[l], tA, tB, stream);
    if (l == 1) run_level_convs<64> (xh[l], pw[l], sshB[l], so[l], tA, tB, stream);
    if (l == 2) run_level_convs<32> (xh[l], pw[l], sshB[l], so[l], tA, tB, stream);
    int Hl = (l == 0) ? 128 : (l == 1) ? 64 : 32;
    int nthr = BATCH * HWs[l];
    k_head_decode<<<(nthr + 255) / 256, 256, 0, stream>>>(
        so[l], boxW[l], boxB[l], clsW[l], clsB[l], boxes, scores,
        Hl, Hl, lvlBase[l], strides[l], sz0[l], sz1[l]);
  }

  // ---- top-k via descending bitonic sort on (score, ~idx) keys ----
  int nk = BATCH * NSORT;
  k_keys<<<nk / 256, 256, 0, stream>>>(scores, keys);
  for (int k = 2; k <= NSORT; k <<= 1)
    for (int j = k >> 1; j > 0; j >>= 1)
      k_bitonic<<<nk / 256, 256, 0, stream>>>(keys, k, j);

  // ---- greedy NMS + output ----
  k_nms<<<BATCH, 1024, 0, stream>>>(keys, boxes, (float*)d_out);
}
